// FlashSVDBlockDA_85529978733244
// MI455X (gfx1250) — matrix operand order repaired and checked by reading, not compiled, not run on hardware
//
#include <hip/hip_runtime.h>
#include <hip/hip_bf16.h>

// ---------------------------------------------------------------------------
// CDNA5 (gfx1250) wave32 WMMA transformer block:
//   low-rank QKV -> flash attention -> low-rank Wo (+res) -> LN1 ->
//   low-rank FFN (GELU) (+res) -> LN2
// All matmuls via v_wmma_f32_16x16x32_f16 (f16 operands, f32 accum).
// Operand loads vectorized to 2 x b128 per WMMA operand.
// ---------------------------------------------------------------------------

typedef __attribute__((ext_vector_type(16))) _Float16     v16h;
typedef __attribute__((ext_vector_type(8)))  float        v8f;
typedef __attribute__((ext_vector_type(4)))  unsigned int u32x4;
typedef __attribute__((ext_vector_type(2)))  unsigned int u32x2;

union F16x16 { v16h v; u32x4 q[2]; unsigned int u[8]; _Float16 h[16]; };
union U2H2   { u32x2 u2; _Float16 h[4]; };

__device__ __forceinline__ int lane_g() { return (threadIdx.x >> 4) & 1; }

__device__ __forceinline__ v8f zero8() {
  v8f z;
#pragma unroll
  for (int i = 0; i < 8; ++i) z[i] = 0.f;
  return z;
}

// A operand (16x32 f16): m = lane&15, k(e) = (e>>3)*16 + g*8 + (e&7).
// Two 16-byte contiguous runs: [g*8 .. g*8+7] and [16+g*8 .. 16+g*8+7].
// rowptr (k-contiguous) must be 16-byte aligned at rowptr + g*8.
__device__ __forceinline__ v16h load_a16(const _Float16* rowptr) {
  F16x16 f;
  const int g = lane_g();
  f.q[0] = *(const u32x4*)(rowptr + (g << 3));
  f.q[1] = *(const u32x4*)(rowptr + 16 + (g << 3));
  return f.v;
}

// B operand (32x16 f16): n = lane&15, k(e) = (e>>3)*16 + g*8 + (e&7).
// Two 16-byte contiguous runs: [g*8 .. g*8+7] and [16+g*8 .. 16+g*8+7].
__device__ __forceinline__ v16h load_b16(const _Float16* colptr) {
  F16x16 f;
  const int g = lane_g();
  f.q[0] = *(const u32x4*)(colptr + (g << 3));
  f.q[1] = *(const u32x4*)(colptr + 16 + (g << 3));
  return f.v;
}

__device__ __forceinline__ v8f wmma16(v16h a, v16h b, v8f c) {
  return __builtin_amdgcn_wmma_f32_16x16x32_f16(false, a, false, b, (short)0, c, false, false);
}

__device__ __forceinline__ float redmax16(float x) {
#pragma unroll
  for (int d = 1; d < 16; d <<= 1) x = fmaxf(x, __shfl_xor(x, d, 16));
  return x;
}
__device__ __forceinline__ float redsum16(float x) {
#pragma unroll
  for (int d = 1; d < 16; d <<= 1) x += __shfl_xor(x, d, 16);
  return x;
}

// ---------------------------------------------------------------------------
// f32 -> f16 convert
// ---------------------------------------------------------------------------
__global__ void cvt_f32_f16_k(const float* __restrict__ s, _Float16* __restrict__ d, int n) {
  int i = blockIdx.x * 256 + threadIdx.x;
  if (i < n) d[i] = (_Float16)s[i];
}

// ---------------------------------------------------------------------------
// Generic tiled WMMA GEMM: C = epilogue(A[Mr x K] * B[K x N] + bias) (+res)
//   block = 128 threads (4 waves); block tile = 64 rows x (16*NT) cols
//   A: f16 row-major (lda), per-z offset aZ.  B: f16 row-major (ldb), per-z bZ.
//   B k-chunk staged through LDS transposed (k-contiguous per column).
//   OUT_TR16: also/instead store transposed f16: CT[z*tZ + col*ldt + row].
// ---------------------------------------------------------------------------
template <int NT, bool HAS_BIAS, bool DO_GELU, bool HAS_RES, bool OUT_F32,
          bool OUT_F16, bool OUT_TR16>
__global__ void __launch_bounds__(128)
gemm_wmma_k(const _Float16* __restrict__ A, long aZ, int lda,
            const _Float16* __restrict__ B, long bZ, int ldb,
            const float* __restrict__ bias, long biasZ,
            const float* __restrict__ res,
            float* __restrict__ C32, _Float16* __restrict__ C16,
            long cZ, int ldc,
            _Float16* __restrict__ CT, long tZ, int ldt,
            int K) {
  constexpr int BN = 16 * NT;
  __shared__ __align__(16) _Float16 Bt[BN][40];  // [n][k], 80B rows (16B aligned)

  const int tid  = threadIdx.x;
  const int wave = tid >> 5;
  const int r16  = tid & 15;
  const int g    = lane_g();
  const int z    = blockIdx.z;

  const _Float16* Ab = A + (long)z * aZ;
  const _Float16* Bb = B + (long)z * bZ;
  const long n0   = (long)blockIdx.x * BN;
  const long mrow = (long)blockIdx.y * 64 + wave * 16 + r16;

  v8f acc[NT];
#pragma unroll
  for (int t = 0; t < NT; ++t) acc[t] = zero8();

  for (int kc = 0; kc < K; kc += 32) {
    // cooperative load of B chunk [32 x BN] into Bt[n][k]
    for (int i = tid * 4; i < 32 * BN; i += 128 * 4) {
      const int k = i / BN, n = i % BN;
      U2H2 d;
      d.u2 = *(const u32x2*)(Bb + (long)(kc + k) * ldb + n0 + n);
      Bt[n + 0][k] = d.h[0];
      Bt[n + 1][k] = d.h[1];
      Bt[n + 2][k] = d.h[2];
      Bt[n + 3][k] = d.h[3];
    }
    if (kc + 32 < K) __builtin_prefetch(Bb + (long)(kc + 32) * ldb + n0, 0, 1);
    __syncthreads();

    const v16h a = load_a16(Ab + mrow * (long)lda + kc);
#pragma unroll
    for (int t = 0; t < NT; ++t) {
      const v16h b = load_b16(&Bt[t * 16 + r16][0]);
      acc[t] = wmma16(a, b, acc[t]);
    }
    __syncthreads();
  }

  const long row0 = (long)blockIdx.y * 64 + wave * 16 + g * 8;
#pragma unroll
  for (int t = 0; t < NT; ++t) {
    const long col = n0 + t * 16 + r16;
    float bval = 0.f;
    if constexpr (HAS_BIAS) bval = bias[(long)z * biasZ + col];
#pragma unroll
    for (int v = 0; v < 8; ++v) {
      float val = acc[t][v] + bval;
      if constexpr (DO_GELU) val = 0.5f * val * (1.f + erff(val * 0.70710678118654752f));
      const long row = row0 + v;
      if constexpr (HAS_RES) val += res[row * (long)ldc + col];
      if constexpr (OUT_F32) C32[(long)z * cZ + row * (long)ldc + col] = val;
      if constexpr (OUT_F16) C16[(long)z * cZ + row * (long)ldc + col] = (_Float16)val;
      if constexpr (OUT_TR16) CT[(long)z * tZ + col * (long)ldt + row] = (_Float16)val;
    }
  }
}

// ---------------------------------------------------------------------------
// Flash attention: one wave per 16-query tile per (b,h); streams K/V in
// 32-key chunks with online softmax.
//   Q,K layout: [B, M, H, DH] f16.   Vt layout: [H, DH, B*M] f16 (transposed).
//   Output attn -> [B, M, H*DH] f16.
// ---------------------------------------------------------------------------
__global__ void __launch_bounds__(128)
flash_attn_k(const _Float16* __restrict__ Q, const _Float16* __restrict__ K,
             const _Float16* __restrict__ Vt, const float* __restrict__ mask,
             _Float16* __restrict__ O) {
  __shared__ __align__(16) _Float16 pt[4][16][40];  // per-wave P tile [m][k]

  const int tid  = threadIdx.x;
  const int wave = tid >> 5;
  const int r16  = tid & 15;
  const int g    = lane_g();
  const int h = blockIdx.y;
  const int b = blockIdx.z;
  const int mtile = blockIdx.x * 4 + wave;
  const long ROWS = 8192;

  const _Float16* Qb = Q + ((long)b * 1024 + mtile * 16 + r16) * 768 + h * 64;
  const v16h qa0 = load_a16(Qb);       // dh 0..31
  const v16h qa1 = load_a16(Qb + 32);  // dh 32..63

  const _Float16* Kb  = K + (long)b * 1024 * 768 + h * 64;
  const _Float16* VtB = Vt + (long)h * 64 * ROWS + (long)b * 1024;
  const float*    mb  = mask + (long)b * 1024;

  float mo[8], l[8];
  v8f acc[4];
#pragma unroll
  for (int v = 0; v < 8; ++v) { mo[v] = -1e30f; l[v] = 0.f; }
#pragma unroll
  for (int t = 0; t < 4; ++t) acc[t] = zero8();

  for (int n0 = 0; n0 < 1024; n0 += 32) {
    // scores S[16 x 32] = Q (16x64) * K^T (64x32), as two 16x16 C tiles
    v8f s0 = zero8(), s1 = zero8();
    {
      const _Float16* kp0 = Kb + (long)(n0 + r16) * 768;
      s0 = wmma16(qa0, load_b16(kp0), s0);
      s0 = wmma16(qa1, load_b16(kp0 + 32), s0);
      const _Float16* kp1 = Kb + (long)(n0 + 16 + r16) * 768;
      s1 = wmma16(qa0, load_b16(kp1), s1);
      s1 = wmma16(qa1, load_b16(kp1 + 32), s1);
    }
    const float mv0 = mb[n0 + r16];
    const float mv1 = mb[n0 + 16 + r16];

#pragma unroll
    for (int v = 0; v < 8; ++v) {
      const float a0 = s0[v] * 0.125f + mv0;  // 1/sqrt(64) = 0.125
      const float a1 = s1[v] * 0.125f + mv1;
      const float tmax = redmax16(fmaxf(a0, a1));      // same row, 32 cols
      const float mn   = fmaxf(mo[v], tmax);
      const float corr = __expf(mo[v] - mn);
      const float p0 = __expf(a0 - mn);
      const float p1 = __expf(a1 - mn);
      l[v] = l[v] * corr + redsum16(p0 + p1);
      mo[v] = mn;
#pragma unroll
      for (int t = 0; t < 4; ++t) acc[t][v] *= corr;
      pt[wave][v + 8 * g][r16]      = (_Float16)p0;  // C layout -> LDS
      pt[wave][v + 8 * g][16 + r16] = (_Float16)p1;
    }

    // P (16x32) as A operand from LDS; acc += P * V (32 x 64), V k-contiguous
    const v16h pa = load_a16(&pt[wave][r16][0]);
#pragma unroll
    for (int t = 0; t < 4; ++t) {
      const v16h bv = load_b16(VtB + (long)(t * 16 + r16) * ROWS + n0);
      acc[t] = wmma16(pa, bv, acc[t]);
    }
  }

#pragma unroll
  for (int t = 0; t < 4; ++t) {
#pragma unroll
    for (int v = 0; v < 8; ++v) {
      const long row = (long)b * 1024 + mtile * 16 + v + 8 * g;
      O[row * 768 + h * 64 + t * 16 + r16] = (_Float16)(acc[t][v] / l[v]);
    }
  }
}

// ---------------------------------------------------------------------------
// Row LayerNorm over 768 cols; writes f32 (always) and f16 (optional).
// ---------------------------------------------------------------------------
__global__ void __launch_bounds__(256)
layernorm_k(const float* __restrict__ X, const float* __restrict__ gw,
            const float* __restrict__ bw, float* __restrict__ O32,
            _Float16* __restrict__ O16) {
  __shared__ float red[2][8];
  const long row = blockIdx.x;
  const float* xr = X + row * 768;
  float s = 0.f, ss = 0.f;
  for (int c = threadIdx.x; c < 768; c += 256) {
    const float v = xr[c];
    s += v; ss += v * v;
  }
#pragma unroll
  for (int d = 1; d < 32; d <<= 1) { s += __shfl_xor(s, d, 32); ss += __shfl_xor(ss, d, 32); }
  if ((threadIdx.x & 31) == 0) { red[0][threadIdx.x >> 5] = s; red[1][threadIdx.x >> 5] = ss; }
  __syncthreads();
  float ts = 0.f, tss = 0.f;
#pragma unroll
  for (int i = 0; i < 8; ++i) { ts += red[0][i]; tss += red[1][i]; }
  const float mu  = ts * (1.f / 768.f);
  const float var = tss * (1.f / 768.f) - mu * mu;
  const float inv = rsqrtf(var + 1e-12f);
  for (int c = threadIdx.x; c < 768; c += 256) {
    const float v = (xr[c] - mu) * inv * gw[c] + bw[c];
    if (O32) O32[row * 768 + c] = v;
    if (O16) O16[row * 768 + c] = (_Float16)v;
  }
}

// ---------------------------------------------------------------------------
// Host launch
// ---------------------------------------------------------------------------
extern "C" void kernel_launch(void* const* d_in, const int* in_sizes, int n_in,
                              void* d_out, int out_size, void* d_ws, size_t ws_size,
                              hipStream_t stream) {
  (void)in_sizes; (void)n_in; (void)out_size; (void)ws_size;
  const int Bn = 8, Mn = 1024, DM = 768, Hh = 12, DH = 64, Rr = 32;
  const int RFF = 384, DFF = 3072, RWO = 384;
  const long ROWS = (long)Bn * Mn;  // 8192

  const float* x    = (const float*)d_in[0];
  const float* mask = (const float*)d_in[1];
  const float* Pq = (const float*)d_in[2];
  const float* Vq = (const float*)d_in[3];
  const float* bq = (const float*)d_in[4];
  const float* Pk = (const float*)d_in[5];
  const float* Vk = (const float*)d_in[6];
  const float* bk = (const float*)d_in[7];
  const float* Pv = (const float*)d_in[8];
  const float* Vv = (const float*)d_in[9];
  const float* bv = (const float*)d_in[10];
  const float* Uo = (const float*)d_in[11];
  const float* Vo = (const float*)d_in[12];
  const float* bo = (const float*)d_in[13];
  const float* U1 = (const float*)d_in[14];
  const float* V1 = (const float*)d_in[15];
  const float* b1 = (const float*)d_in[16];
  const float* U2 = (const float*)d_in[17];
  const float* V2 = (const float*)d_in[18];
  const float* b2 = (const float*)d_in[19];
  const float* ln1g = (const float*)d_in[20];
  const float* ln1b = (const float*)d_in[21];
  const float* ln2g = (const float*)d_in[22];
  const float* ln2b = (const float*)d_in[23];

  char* wp = (char*)d_ws;
  auto alloc = [&](size_t bytes) -> char* {
    char* r = wp;
    wp += (bytes + 255) & ~(size_t)255;
    return r;
  };

  _Float16* Xh  = (_Float16*)alloc(ROWS * DM * 2);   // x f16; reused for attn out
  _Float16* Qf  = (_Float16*)alloc(ROWS * DM * 2);   // [B,M,H,DH]
  _Float16* Kf  = (_Float16*)alloc(ROWS * DM * 2);   // [B,M,H,DH]
  _Float16* Vtf = (_Float16*)alloc(ROWS * DM * 2);   // [H,DH,B*M] (transposed)
  _Float16* Tf  = (_Float16*)alloc(ROWS * 384 * 2);  // rank-384 scratch (reused)
  float*    h1  = (float*)   alloc(ROWS * DM * 4);   // pre-LN buffer (reused)
  float*    x1f = (float*)   alloc(ROWS * DM * 4);   // x1 f32 (residual)
  _Float16* x1h = (_Float16*)alloc(ROWS * DM * 2);   // x1 f16 (GEMM input)
  _Float16* hdn = (_Float16*)alloc(ROWS * DFF * 2);  // FFN hidden f16

  _Float16* wPq = (_Float16*)alloc((size_t)Hh * DM * Rr * 2);
  _Float16* wVq = (_Float16*)alloc((size_t)Hh * Rr * DH * 2);
  _Float16* wPk = (_Float16*)alloc((size_t)Hh * DM * Rr * 2);
  _Float16* wVk = (_Float16*)alloc((size_t)Hh * Rr * DH * 2);
  _Float16* wPv = (_Float16*)alloc((size_t)Hh * DM * Rr * 2);
  _Float16* wVv = (_Float16*)alloc((size_t)Hh * Rr * DH * 2);
  _Float16* wUo = (_Float16*)alloc((size_t)DM * RWO * 2);
  _Float16* wVo = (_Float16*)alloc((size_t)RWO * DM * 2);
  _Float16* wU1 = (_Float16*)alloc((size_t)DM * RFF * 2);
  _Float16* wV1 = (_Float16*)alloc((size_t)RFF * DFF * 2);
  _Float16* wU2 = (_Float16*)alloc((size_t)DFF * RFF * 2);
  _Float16* wV2 = (_Float16*)alloc((size_t)RFF * DM * 2);

  auto cvt = [&](const float* s, _Float16* d, long n) {
    cvt_f32_f16_k<<<dim3((unsigned)((n + 255) / 256)), dim3(256), 0, stream>>>(s, d, (int)n);
  };
  cvt(x, Xh, ROWS * DM);
  cvt(Pq, wPq, (long)Hh * DM * Rr); cvt(Vq, wVq, (long)Hh * Rr * DH);
  cvt(Pk, wPk, (long)Hh * DM * Rr); cvt(Vk, wVk, (long)Hh * Rr * DH);
  cvt(Pv, wPv, (long)Hh * DM * Rr); cvt(Vv, wVv, (long)Hh * Rr * DH);
  cvt(Uo, wUo, (long)DM * RWO);     cvt(Vo, wVo, (long)RWO * DM);
  cvt(U1, wU1, (long)DM * RFF);     cvt(V1, wV1, (long)RFF * DFF);
  cvt(U2, wU2, (long)DFF * RFF);    cvt(V2, wV2, (long)RFF * DM);

  const dim3 blk(128);
  const unsigned MG = (unsigned)(ROWS / 64);  // 128 row-tiles

  // ---- QKV stage1 (x@P -> T[B,M,H,R]) then stage2 (T@V + b) per head ----
  auto qkv_s1 = [&](const _Float16* P) {
    gemm_wmma_k<2, false, false, false, false, true, false>
        <<<dim3(1, MG, Hh), blk, 0, stream>>>(
            Xh, 0, DM, P, (long)DM * Rr, Rr, nullptr, 0, nullptr,
            nullptr, Tf, Rr, Hh * Rr, nullptr, 0, 0, DM);
  };
  // Q, K: normal [B,M,H,DH] layout
  qkv_s1(wPq);
  gemm_wmma_k<4, true, false, false, false, true, false>
      <<<dim3(1, MG, Hh), blk, 0, stream>>>(
          Tf, Rr, Hh * Rr, wVq, (long)Rr * DH, DH, bq, DH, nullptr,
          nullptr, Qf, DH, DM, nullptr, 0, 0, Rr);
  qkv_s1(wPk);
  gemm_wmma_k<4, true, false, false, false, true, false>
      <<<dim3(1, MG, Hh), blk, 0, stream>>>(
          Tf, Rr, Hh * Rr, wVk, (long)Rr * DH, DH, bk, DH, nullptr,
          nullptr, Kf, DH, DM, nullptr, 0, 0, Rr);
  // V: store transposed -> Vt[h][dh][row]
  qkv_s1(wPv);
  gemm_wmma_k<4, true, false, false, false, false, true>
      <<<dim3(1, MG, Hh), blk, 0, stream>>>(
          Tf, Rr, Hh * Rr, wVv, (long)Rr * DH, DH, bv, DH, nullptr,
          nullptr, nullptr, 0, 0, Vtf, (long)DH * ROWS, (int)ROWS, Rr);

  // ---- flash attention -> attn f16 (reuse Xh) ----
  flash_attn_k<<<dim3(Mn / 16 / 4, Hh, Bn), blk, 0, stream>>>(Qf, Kf, Vtf, mask, Xh);

  // ---- Wo: attn @ Uo -> Tf ; Tf @ Vo + bo + x -> h1 (f32) ----
  gemm_wmma_k<4, false, false, false, false, true, false>
      <<<dim3(RWO / 64, MG, 1), blk, 0, stream>>>(
          Xh, 0, DM, wUo, 0, RWO, nullptr, 0, nullptr,
          nullptr, Tf, 0, RWO, nullptr, 0, 0, DM);
  gemm_wmma_k<4, true, false, true, true, false, false>
      <<<dim3(DM / 64, MG, 1), blk, 0, stream>>>(
          Tf, 0, RWO, wVo, 0, DM, bo, 0, x,
          h1, nullptr, 0, DM, nullptr, 0, 0, RWO);

  // ---- LN1 -> x1 (f32 + f16) ----
  layernorm_k<<<dim3((unsigned)ROWS), dim3(256), 0, stream>>>(h1, ln1g, ln1b, x1f, x1h);

  // ---- FFN ----
  gemm_wmma_k<4, false, false, false, false, true, false>
      <<<dim3(RFF / 64, MG, 1), blk, 0, stream>>>(
          x1h, 0, DM, wU1, 0, RFF, nullptr, 0, nullptr,
          nullptr, Tf, 0, RFF, nullptr, 0, 0, DM);
  gemm_wmma_k<4, true, true, false, false, true, false>
      <<<dim3(DFF / 64, MG, 1), blk, 0, stream>>>(
          Tf, 0, RFF, wV1, 0, DFF, b1, 0, nullptr,
          nullptr, hdn, 0, DFF, nullptr, 0, 0, RFF);
  gemm_wmma_k<4, false, false, false, false, true, false>
      <<<dim3(RFF / 64, MG, 1), blk, 0, stream>>>(
          hdn, 0, DFF, wU2, 0, RFF, nullptr, 0, nullptr,
          nullptr, Tf, 0, RFF, nullptr, 0, 0, DFF);
  gemm_wmma_k<4, true, false, true, true, false, false>
      <<<dim3(DM / 64, MG, 1), blk, 0, stream>>>(
          Tf, 0, RFF, wV2, 0, DM, b2, 0, x1f,
          h1, nullptr, 0, DM, nullptr, 0, 0, RFF);

  // ---- LN2 -> d_out (f32) ----
  layernorm_k<<<dim3((unsigned)ROWS), dim3(256), 0, stream>>>(h1, ln2g, ln2b,
                                                              (float*)d_out, nullptr);
}
